// L2XModel_3487513445110
// MI455X (gfx1250) — matrix operand order repaired and checked by reading, not compiled, not run on hardware
//
#include <hip/hip_runtime.h>
#include <hip/hip_bf16.h>
#include <stdint.h>

// ---------------- types ----------------
typedef __bf16 bf16_t;
typedef __attribute__((ext_vector_type(16))) __bf16 v16bf;
typedef __attribute__((ext_vector_type(8)))  float  v8f;

union Frag { v16bf v; uint4 q[2]; };

#define EPS_F 1.1920929e-07f
#define INV_TEMP (1.0f / 0.3f)

// problem constants (reference is fixed-shape)
#define BB 512
#define LL 512
#define KK 10
#define CR 100     // real channel count (E=F=H=100)
#define CP 128     // padded channel count (multiple of 32 for WMMA K-loop)

// LDS input tile for conv kernels: 66 l-rows (64 + 2 halo) x CP channels,
// row stride padded to 136 bf16 (272 B = 68 dwords) so 16-lane x 16B DS reads
// stride banks by 4 -> full 64-bank utilization, 2-cycle optimum.
#define TROW 136
#define TNR  66

// ---------------- weight padding ----------------
// w: [Cout][Cin][3] f32  ->  wp: [3][CP][CP] bf16, zero padded
__global__ __launch_bounds__(256) void wpad3_kernel(const float* __restrict__ w,
                                                    bf16_t* __restrict__ wp) {
    int idx = blockIdx.x * 256 + threadIdx.x;
    if (idx >= 3 * CP * CP) return;
    int dk = idx / (CP * CP);
    int co = (idx / CP) % CP;
    int ci = idx % CP;
    float v = (co < CR && ci < CR) ? w[(co * CR + ci) * 3 + dk] : 0.f;
    wp[idx] = (bf16_t)v;
}

// conv3_w: [100][200] f32 -> wp: [128][256] bf16 ; ci<128 = g-part, ci>=128 = loc-part
__global__ __launch_bounds__(256) void wpadc3_kernel(const float* __restrict__ w,
                                                     bf16_t* __restrict__ wp) {
    int idx = blockIdx.x * 256 + threadIdx.x;
    if (idx >= CP * 256) return;
    int co = idx / 256;
    int ci = idx % 256;
    float v = 0.f;
    if (co < CR) {
        if (ci < CR)                       v = w[co * 200 + ci];
        else if (ci >= CP && ci < CP + CR) v = w[co * 200 + CR + (ci - CP)];
    }
    wp[idx] = (bf16_t)v;
}

// ---------------- embedding gather -> bf16 [B][L][CP] ----------------
__global__ __launch_bounds__(CP) void gather_kernel(const int* __restrict__ x,
                                                    const float* __restrict__ emb,
                                                    bf16_t* __restrict__ E) {
    int pos = blockIdx.x;         // b*L + l
    int c   = threadIdx.x;        // 0..127
    int tok = x[pos];
    float v = (c < CR) ? emb[(size_t)tok * CR + c] : 0.f;
    E[(size_t)pos * CP + c] = (bf16_t)v;
}

// ---------------- k=3 conv as WMMA GEMM (async-LDS staged) ----------------
// IN:  [B][L][CP] bf16 (channel-contiguous), W: [3][CP][CP] bf16, bias: [CR] f32
// OUT: [B][L][CP] bf16, ReLU applied, pad channels forced to zero.
// Block: 256 thr = 8 waves; wave w => cout tile w; block covers 64 l-columns.
// Input tile staged via global_load_async_to_lds_b128 (ASYNCcnt), B fragments
// then read from LDS (halo rows pre-zeroed -> branch-free inner loop).
__global__ __launch_bounds__(256) void conv3x_gemm_kernel(const bf16_t* __restrict__ IN,
                                                          const bf16_t* __restrict__ W,
                                                          const float*  __restrict__ bias,
                                                          bf16_t* __restrict__ OUT) {
    __shared__ __align__(16) bf16_t tile[TNR * TROW];

    const int tid  = threadIdx.x;
    const int wave = tid >> 5;
    const int lane = tid & 31;
    const int g    = lane >> 4;   // half-wave select
    const int ln   = lane & 15;

    const int ltiles = LL >> 6;            // 64-wide l blocks
    const int b  = blockIdx.x / ltiles;
    const int l0 = (blockIdx.x % ltiles) << 6;
    const int co0 = wave << 4;             // 8 waves x 16 = 128 cout

    // zero halo rows (row 0 / row 65) when outside [0, L) -- one elem per thread
    {
        int which = tid >> 7;              // 0 -> row 0, 1 -> row 65
        int c     = tid & (CP - 1);
        int row   = which ? (TNR - 1) : 0;
        int l     = l0 - 1 + row;
        if (l < 0 || l >= LL) tile[row * TROW + c] = (bf16_t)0.f;
    }

    // async-stage input tile: rows [l0-1, l0+64], 16 x 16B segments per row
    for (int i = tid; i < TNR * 16; i += 256) {
        int row = i >> 4;
        int seg = i & 15;                  // 8 channels per segment
        int l   = l0 - 1 + row;
        if (l >= 0 && l < LL) {
            const bf16_t* gp = IN + ((size_t)b * LL + l) * CP + seg * 8;
            uint32_t lds_off = (uint32_t)(uintptr_t)(&tile[row * TROW + seg * 8]);
            asm volatile("global_load_async_to_lds_b128 %0, %1, off"
                         :: "v"(lds_off), "v"(gp) : "memory");
        }
    }
    asm volatile("s_wait_asynccnt 0x0" ::: "memory");
    __syncthreads();

    const v8f zero = {0.f,0.f,0.f,0.f,0.f,0.f,0.f,0.f};
    v8f acc[4] = {zero, zero, zero, zero};

    for (int dk = 0; dk < 3; ++dk) {
        const bf16_t* Wd = W + (size_t)dk * CP * CP;
        for (int kc = 0; kc < CP; kc += 32) {
            // A fragment: row M = co0+ln (both halves), K per documented layout
            Frag a;
            const bf16_t* ap = Wd + (size_t)(co0 + ln) * CP + kc + g * 8;
            a.q[0] = *(const uint4*)(ap);
            a.q[1] = *(const uint4*)(ap + 16);
            for (int t = 0; t < 4; ++t) {
                // column n = ln; LDS row = (l - (l0-1)) = t*16 + ln + dk
                const bf16_t* bp = &tile[(t * 16 + ln + dk) * TROW + kc + g * 8];
                Frag bfr;
                bfr.q[0] = *(const uint4*)(bp);
                bfr.q[1] = *(const uint4*)(bp + 16);
                acc[t] = __builtin_amdgcn_wmma_f32_16x16x32_bf16(
                    false, a.v, false, bfr.v, (short)0, acc[t], false, false);
            }
        }
    }

    // store: lane holds column n=ln, rows cbase..cbase+7 (contiguous channels)
    const int cbase = co0 + g * 8;
    for (int t = 0; t < 4; ++t) {
        int l = l0 + t * 16 + ln;
        union { uint4 q; bf16_t h[8]; } st;
        for (int r = 0; r < 8; ++r) {
            int co = cbase + r;
            float v = 0.f;
            if (co < CR) {
                v = acc[t][r] + bias[co];
                v = v > 0.f ? v : 0.f;
            }
            st.h[r] = (bf16_t)v;
        }
        *(uint4*)(OUT + ((size_t)b * LL + l) * CP + cbase) = st.q;
    }
}

// ---------------- global pool + gi MLP -> g bf16 [B][CP] ----------------
__global__ __launch_bounds__(CP) void gpool_kernel(const bf16_t* __restrict__ H,
                                                   const float* __restrict__ gi_w,
                                                   const float* __restrict__ gi_b,
                                                   bf16_t* __restrict__ G) {
    __shared__ float hm[CP];
    int b = blockIdx.x, c = threadIdx.x;
    float s = 0.f;
    for (int l = 0; l < LL; ++l)
        s += (float)H[((size_t)b * LL + l) * CP + c];
    hm[c] = s * (1.f / (float)LL);
    __syncthreads();
    float o = 0.f;
    if (c < CR) {
        float a = gi_b[c];
        for (int f = 0; f < CR; ++f) a += gi_w[c * CR + f] * hm[f];
        o = a > 0.f ? a : 0.f;
    }
    G[(size_t)b * CP + c] = (bf16_t)o;
}

// ---------------- fused conv3 (k=1, 256->128) + conv4 (128->1) -> logits ----------------
// Block: 256 thr = 8 waves over cout; block covers 16 l-columns; K = 256 (g | loc).
__global__ __launch_bounds__(256) void conv34_kernel(const bf16_t* __restrict__ GV,   // [B][CP]
                                                     const bf16_t* __restrict__ LOC,  // [B][L][CP]
                                                     const bf16_t* __restrict__ W3,   // [CP][256]
                                                     const float* __restrict__ b3,
                                                     const float* __restrict__ w4,
                                                     const float* __restrict__ b4,
                                                     float* __restrict__ LOGITS) {
    __shared__ float red[16];
    const int tid = threadIdx.x, wave = tid >> 5, lane = tid & 31;
    const int g = lane >> 4, ln = lane & 15;
    const int ltiles = LL >> 4;
    const int b  = blockIdx.x / ltiles;
    const int l0 = (blockIdx.x % ltiles) << 4;
    const int co0 = wave << 4;

    if (tid < 16) red[tid] = b4[0];
    __syncthreads();

    const v8f zero = {0.f,0.f,0.f,0.f,0.f,0.f,0.f,0.f};
    v8f acc = zero;
    for (int kc = 0; kc < 256; kc += 32) {
        Frag a;
        const bf16_t* ap = W3 + (size_t)(co0 + ln) * 256 + kc + g * 8;
        a.q[0] = *(const uint4*)(ap);
        a.q[1] = *(const uint4*)(ap + 16);
        Frag bfr;
        int ci = kc + g * 8;
        const bf16_t* bp = (ci < CP)
            ? (GV + (size_t)b * CP + ci)                                  // broadcast g over l
            : (LOC + ((size_t)b * LL + l0 + ln) * CP + (ci - CP));
        bfr.q[0] = *(const uint4*)(bp);
        bfr.q[1] = *(const uint4*)(bp + 16);
        acc = __builtin_amdgcn_wmma_f32_16x16x32_bf16(
            false, a.v, false, bfr.v, (short)0, acc, false, false);
    }

    // conv4: weighted reduction over cout, fused with ReLU+bias of conv3
    float part = 0.f;
    const int cbase = co0 + g * 8;
    for (int r = 0; r < 8; ++r) {
        int co = cbase + r;
        if (co < CR) {
            float v = acc[r] + b3[co];
            v = v > 0.f ? v : 0.f;
            part += v * w4[co];
        }
    }
    atomicAdd(&red[ln], part);   // ds_add_f32 in LDS
    __syncthreads();
    if (tid < 16) LOGITS[(size_t)b * LL + l0 + tid] = red[tid];
}

// ---------------- gumbel top-k continuous sampling -> csamples [B][L] ----------------
__global__ __launch_bounds__(256) void gumbel_kernel(const float* __restrict__ u,
                                                     const float* __restrict__ logits,
                                                     float* __restrict__ cs) {
    __shared__ float sm[LL];
    __shared__ float mx[LL];
    __shared__ float red[256];
    const int b = blockIdx.x, tid = threadIdx.x;
    mx[tid] = 0.f; mx[tid + 256] = 0.f;

    for (int k = 0; k < KK; ++k) {
        for (int i = tid; i < LL; i += 256) {
            float uc = u[((size_t)b * KK + k) * LL + i];
            uc = fminf(fmaxf(uc, EPS_F), 1.f - EPS_F);
            float gum = -logf(-logf(uc));
            sm[i] = (gum + logits[(size_t)b * LL + i]) * INV_TEMP;
        }
        __syncthreads();
        // max-reduce over L
        red[tid] = fmaxf(sm[tid], sm[tid + 256]);
        __syncthreads();
        for (int s = 128; s > 0; s >>= 1) {
            if (tid < s) red[tid] = fmaxf(red[tid], red[tid + s]);
            __syncthreads();
        }
        float gm = red[0];
        __syncthreads();
        // sum(exp)
        float e0 = expf(sm[tid] - gm), e1 = expf(sm[tid + 256] - gm);
        red[tid] = e0 + e1;
        __syncthreads();
        for (int s = 128; s > 0; s >>= 1) {
            if (tid < s) red[tid] += red[tid + s];
            __syncthreads();
        }
        float inv = 1.f / red[0];
        __syncthreads();
        mx[tid]       = fmaxf(mx[tid],       e0 * inv);
        mx[tid + 256] = fmaxf(mx[tid + 256], e1 * inv);
        __syncthreads();
    }
    cs[(size_t)b * LL + tid]       = mx[tid];
    cs[(size_t)b * LL + tid + 256] = mx[tid + 256];
}

// ---------------- masked mean pool + fc1 + sigmoid head ----------------
__global__ __launch_bounds__(CP) void head_kernel(const bf16_t* __restrict__ E,
                                                  const float* __restrict__ cs,
                                                  const float* __restrict__ fc1_w,
                                                  const float* __restrict__ fc1_b,
                                                  const float* __restrict__ head_w,
                                                  const float* __restrict__ head_b,
                                                  float* __restrict__ out) {
    __shared__ float csl[LL];
    __shared__ float pooled[CP];
    __shared__ float hv[CR];
    const int b = blockIdx.x, tid = threadIdx.x;   // 128 threads
    for (int i = tid; i < LL; i += CP) csl[i] = cs[(size_t)b * LL + i];
    __syncthreads();
    float s = 0.f;
    for (int l = 0; l < LL; ++l)
        s += (float)E[((size_t)b * LL + l) * CP + tid] * csl[l];
    pooled[tid] = s * (1.f / (float)LL);
    __syncthreads();
    if (tid < CR) {
        float a = fc1_b[tid];
        for (int e = 0; e < CR; ++e) a += fc1_w[tid * CR + e] * pooled[e];
        hv[tid] = a > 0.f ? a : 0.f;
    }
    __syncthreads();
    if (tid == 0) {
        float a = head_b[0];
        for (int h = 0; h < CR; ++h) a += head_w[h] * hv[h];
        out[b] = 1.f / (1.f + expf(-a));
    }
}

// ---------------- launcher ----------------
extern "C" void kernel_launch(void* const* d_in, const int* in_sizes, int n_in,
                              void* d_out, int out_size, void* d_ws, size_t ws_size,
                              hipStream_t stream) {
    const int*   x       = (const int*)  d_in[0];
    const float* u       = (const float*)d_in[1];
    const float* emb     = (const float*)d_in[2];
    const float* conv1_w = (const float*)d_in[3];
    const float* conv1_b = (const float*)d_in[4];
    const float* gi_w    = (const float*)d_in[5];
    const float* gi_b    = (const float*)d_in[6];
    const float* conv2_w = (const float*)d_in[7];
    const float* conv2_b = (const float*)d_in[8];
    const float* li_w    = (const float*)d_in[9];
    const float* li_b    = (const float*)d_in[10];
    const float* conv3_w = (const float*)d_in[11];
    const float* conv3_b = (const float*)d_in[12];
    const float* conv4_w = (const float*)d_in[13];
    const float* conv4_b = (const float*)d_in[14];
    const float* fc1_w   = (const float*)d_in[15];
    const float* fc1_b   = (const float*)d_in[16];
    const float* head_w  = (const float*)d_in[17];
    const float* head_b  = (const float*)d_in[18];

    float* out_pred = (float*)d_out;            // [B]
    float* cs       = (float*)d_out + BB;       // [B][L]

    // workspace carve-out (256B aligned slabs)
    char* p = (char*)d_ws;
    auto carve = [&](size_t bytes) {
        void* r = (void*)p;
        p += (bytes + 255) & ~(size_t)255;
        return r;
    };
    const size_t act_bytes = (size_t)BB * LL * CP * sizeof(bf16_t);
    bf16_t* e_bf  = (bf16_t*)carve(act_bytes);
    bf16_t* h_bf  = (bf16_t*)carve(act_bytes);     // also reused as loc
    bf16_t* c2_bf = (bf16_t*)carve(act_bytes);
    bf16_t* g_bf  = (bf16_t*)carve((size_t)BB * CP * sizeof(bf16_t));
    float*  lg    = (float*) carve((size_t)BB * LL * sizeof(float));
    bf16_t* Wb1   = (bf16_t*)carve((size_t)3 * CP * CP * sizeof(bf16_t));
    bf16_t* Wb2   = (bf16_t*)carve((size_t)3 * CP * CP * sizeof(bf16_t));
    bf16_t* Wb3   = (bf16_t*)carve((size_t)3 * CP * CP * sizeof(bf16_t));
    bf16_t* W3p   = (bf16_t*)carve((size_t)CP * 256 * sizeof(bf16_t));
    bf16_t* loc_bf = h_bf;   // alias: h dead after gpool+conv2

    // 1) weight padding / bf16 conversion
    {
        int n3 = 3 * CP * CP;
        wpad3_kernel<<<(n3 + 255) / 256, 256, 0, stream>>>(conv1_w, Wb1);
        wpad3_kernel<<<(n3 + 255) / 256, 256, 0, stream>>>(conv2_w, Wb2);
        wpad3_kernel<<<(n3 + 255) / 256, 256, 0, stream>>>(li_w,    Wb3);
        int nc3 = CP * 256;
        wpadc3_kernel<<<(nc3 + 255) / 256, 256, 0, stream>>>(conv3_w, W3p);
    }

    // 2) embedding gather -> bf16 [B][L][128]
    gather_kernel<<<BB * LL, CP, 0, stream>>>(x, emb, e_bf);

    // 3) conv1 (WMMA, async-LDS staged) -> h
    const int gemm_blocks = BB * (LL / 64);
    conv3x_gemm_kernel<<<gemm_blocks, 256, 0, stream>>>(e_bf, Wb1, conv1_b, h_bf);

    // 4) global pool + gi MLP -> g
    gpool_kernel<<<BB, CP, 0, stream>>>(h_bf, gi_w, gi_b, g_bf);

    // 5) conv2 (WMMA) -> c2
    conv3x_gemm_kernel<<<gemm_blocks, 256, 0, stream>>>(h_bf, Wb2, conv2_b, c2_bf);

    // 6) li conv (WMMA) -> loc (aliases h buffer)
    conv3x_gemm_kernel<<<gemm_blocks, 256, 0, stream>>>(c2_bf, Wb3, li_b, loc_bf);

    // 7) fused conv3+conv4 (WMMA + LDS ds_add_f32 reduce) -> logits
    conv34_kernel<<<BB * (LL / 16), 256, 0, stream>>>(g_bf, loc_bf, W3p, conv3_b,
                                                      conv4_w, conv4_b, lg);

    // 8) gumbel softmax, max over K -> csamples (second output region)
    gumbel_kernel<<<BB, 256, 0, stream>>>(u, lg, cs);

    // 9) masked mean pool + fc1 + sigmoid head -> out (first output region)
    head_kernel<<<BB, CP, 0, stream>>>(e_bf, cs, fc1_w, fc1_b, head_w, head_b, out_pred);
}
